// KGANEncoder_71425306133083
// MI455X (gfx1250) — matrix operand order repaired and checked by reading, not compiled, not run on hardware
//
#include <hip/hip_runtime.h>
#include <hip/hip_bf16.h>

typedef __attribute__((ext_vector_type(16))) _Float16 v16h;
typedef __attribute__((ext_vector_type(8)))  float    v8f;

#define WMMA_F16(a_, b_, c_) \
  __builtin_amdgcn_wmma_f32_16x16x32_f16(false, (a_), false, (b_), (short)0, (c_), false, false)

// ---------- helpers ----------
__device__ __forceinline__ float wave_sum32(float v) {
  for (int m = 16; m >= 1; m >>= 1) v += __shfl_xor(v, m, 32);
  return v;
}
__device__ __forceinline__ float half_sum16(float v) {  // reduce within 16-lane halves
  for (int m = 8; m >= 1; m >>= 1) v += __shfl_xor(v, m, 32);
  return v;
}
__device__ __forceinline__ float block_sum128(float v, float* red) {
  int lane = threadIdx.x & 31, wid = threadIdx.x >> 5;
  v = wave_sum32(v);
  __syncthreads();
  if (lane == 0) red[wid] = v;
  __syncthreads();
  return red[0] + red[1] + red[2] + red[3];
}
__device__ __forceinline__ float renorm_fac(float ss) {
  return fminf(1.f, 1.f / (sqrtf(ss) + 1e-7f));
}
__device__ __forceinline__ float leakyf(float x) { return x > 0.f ? x : 0.2f * x; }

// A fragment (16x32 f16, M x K) from an f32 row pointer.
// lanes 0-15: K = kc+{0..15}; lanes 16-31: K = kc+{16..31}; M = lane&15
__device__ __forceinline__ v16h load_a_f32(const float* row, int kc, int lane) {
  int kb = kc + ((lane & 16) ? 16 : 0);
  const float4* p0 = (const float4*)(row + kb);
  const float4* p1 = (const float4*)(row + kb + 8);
  float4 q0 = p0[0], q1 = p0[1], q2 = p1[0], q3 = p1[1];
  v16h a;
  a[0]=(_Float16)q0.x; a[1]=(_Float16)q0.y; a[2]=(_Float16)q0.z; a[3]=(_Float16)q0.w;
  a[4]=(_Float16)q1.x; a[5]=(_Float16)q1.y; a[6]=(_Float16)q1.z; a[7]=(_Float16)q1.w;
  a[8]=(_Float16)q2.x; a[9]=(_Float16)q2.y; a[10]=(_Float16)q2.z; a[11]=(_Float16)q2.w;
  a[12]=(_Float16)q3.x; a[13]=(_Float16)q3.y; a[14]=(_Float16)q3.z; a[15]=(_Float16)q3.w;
  return a;
}
// A fragment from an LDS f16 tile (row stride lda halves)
__device__ __forceinline__ v16h load_a_lds(const _Float16* aL, int lda, int kc, int lane) {
  const _Float16* r = aL + (lane & 15) * lda + kc + ((lane & 16) ? 16 : 0);
  v16h a;
  for (int t = 0; t < 8; t++) { a[t] = r[t]; a[8 + t] = r[8 + t]; }
  return a;
}
// B fragment (32x16 f16, K x N) from weights stored [N][ldw] f16 (i.e. B[k][n] = W[n][k]).
// lane: n = nbase + (lane&15); halves hold K = kc + (lane<16?0:16) + j
__device__ __forceinline__ v16h load_b_w(const _Float16* W, int ldw, int n, int kc, int lane) {
  return *(const v16h*)(W + (size_t)n * ldw + kc + ((lane & 16) ? 16 : 0));
}

// ---------- f32 -> f16 weight convert ----------
__global__ void k_cvt(const float* __restrict__ s, _Float16* __restrict__ d, int n) {
  int i = blockIdx.x * 256 + threadIdx.x;
  if (i < n) d[i] = (_Float16)s[i];
}

// ---------- relation projection: rproj[rel] = A1_right @ renorm(relation_embs[rel]) ----------
__global__ void k_rproj(const float* __restrict__ remb, const float* __restrict__ A1,
                        float* __restrict__ rproj) {
  __shared__ float red[4];
  __shared__ float rr[128];
  int rel = blockIdx.x, d = threadIdx.x;
  float x = remb[rel * 128 + d];
  float ss = block_sum128(x * x, red);
  rr[d] = x * renorm_fac(ss);
  __syncthreads();
  float acc = 0.f;
  const float* a1row = A1 + (size_t)d * 256 + 128;  // right half of A1 row d
  for (int j = 0; j < 128; j++) acc += a1row[j] * rr[j];
  rproj[rel * 128 + d] = acc;
}

// ---------- gather/renorm: h, t1 (renormed), s1 = sum(t1); also write h into output ----------
__global__ void k0_gather(const int* __restrict__ entity_idx, const int* __restrict__ adj_entity,
                          const float* __restrict__ eemb, float* __restrict__ anchors,
                          float* __restrict__ t1, int* __restrict__ ents1w,
                          float* __restrict__ out) {
  __shared__ float red[4];
  __shared__ int e1s[16];
  int b = blockIdx.x, d = threadIdx.x;
  int ei = entity_idx[b];
  if (d < 16) { int e = adj_entity[(size_t)ei * 16 + d]; e1s[d] = e; ents1w[b * 16 + d] = e; }
  float x = eemb[(size_t)ei * 128 + d];
  float ss = block_sum128(x * x, red);   // barriers inside also publish e1s
  float h = x * renorm_fac(ss);
  anchors[(size_t)b * 128 + d] = h;                 // iter-0 anchor
  out[(size_t)b * 640 + 512 + d] = h;               // output h columns
  float s1 = 0.f;
  for (int k = 0; k < 16; k++) {
    float y = eemb[(size_t)e1s[k] * 128 + d];
    float s2 = block_sum128(y * y, red);
    float t = y * renorm_fac(s2);
    t1[((size_t)b * 16 + k) * 128 + d] = t;
    s1 += t;
  }
  anchors[(size_t)(2048 + b) * 128 + d] = s1;       // iter-1 anchor
}

// ---------- generic 16x16-tile WMMA GEMM: Y[m][col0+n] = act(X@W^T + bias) (+=Y opt) ----------
// X: [M][ldx] f32, W: [N][ldw] f16. grid = (M/16, N/64), block = 128 (4 waves, 1 n-tile each)
__global__ void gemm16(const float* __restrict__ X, int ldx, const _Float16* __restrict__ W,
                       int ldw, const float* __restrict__ bias, float* __restrict__ Y, int ldy,
                       int col0, int K, int act, int accum) {
  int lane = threadIdx.x & 31, wid = threadIdx.x >> 5;
  int mbase = blockIdx.x * 16;
  int nbase = (blockIdx.y * 4 + wid) * 16;
  const float* xrow = X + (size_t)(mbase + (lane & 15)) * ldx;
  int n = nbase + (lane & 15);
  v8f c; for (int i = 0; i < 8; i++) c[i] = 0.f;
  for (int kc = 0; kc < K; kc += 32) {
    v16h a = load_a_f32(xrow, kc, lane);
    v16h bf = load_b_w(W, ldw, n, kc, lane);
    c = WMMA_F16(a, bf, c);
  }
  float bv = bias ? bias[n] : 0.f;
  int mo = (lane & 16) ? 8 : 0;
  for (int v = 0; v < 8; v++) {
    float r = c[v] + bv;
    if (act) r = leakyf(r);
    float* yp = Y + (size_t)(mbase + mo + v) * ldy + col0 + n;
    if (accum) r += *yp;
    *yp = r;
  }
}

// ---------- score table: sigmoid(A3 . relu(A2 @ relu(ap[iter,b] + rproj[rel]))) ----------
// rows g = (iter*2048 + b)*32 + rel, 16 rows per block. block = 128 (4 waves x 2 n-tiles)
__global__ void k_score(const float* __restrict__ ap, const float* __restrict__ rproj,
                        const _Float16* __restrict__ A2h, const float* __restrict__ A3,
                        float* __restrict__ score) {
  __shared__ _Float16 aL[16 * 136];
  __shared__ float sm[16];
  int tid = threadIdx.x, lane = tid & 31, wid = tid >> 5;
  int g0 = blockIdx.x * 16;
  int iter = g0 >> 16;
  int b = (g0 >> 5) & 2047;
  int relbase = g0 & 31;
  const float* aprow = ap + (size_t)(iter * 2048 + b) * 128;
  for (int e = tid; e < 16 * 128; e += 128) {
    int m = e >> 7, k = e & 127;
    float v = aprow[k] + rproj[(relbase + m) * 128 + k];
    aL[m * 136 + k] = (_Float16)fmaxf(v, 0.f);     // relu(a)
  }
  if (tid < 16) sm[tid] = 0.f;
  __syncthreads();
  for (int t = 0; t < 2; t++) {
    int nbase = (wid + t * 4) * 16;
    int n = nbase + (lane & 15);
    v8f c; for (int i = 0; i < 8; i++) c[i] = 0.f;
    for (int kc = 0; kc < 128; kc += 32) {
      v16h a = load_a_lds(aL, 136, kc, lane);
      v16h bf = load_b_w(A2h, 128, n, kc, lane);
      c = WMMA_F16(a, bf, c);
    }
    float a3 = A3[n];
    for (int v = 0; v < 8; v++) {
      float val = fmaxf(c[v], 0.f) * a3;           // relu(a2) * A3
      val = half_sum16(val);
      if ((lane & 15) == 0) atomicAdd(&sm[v + ((lane & 16) ? 8 : 0)], val);
    }
  }
  __syncthreads();
  if (tid < 16) score[g0 + tid] = 1.f / (1.f + __expf(-sm[tid]));
}

// ---------- per-b softmax + weighted neighbor sums (s0 from t1, s1w from hop-2 gathers) ----------
__global__ void k4_softmax_ws(const int* __restrict__ entity_idx, const int* __restrict__ adj_entity,
                              const int* __restrict__ adj_relation, const int* __restrict__ ents1w,
                              const float* __restrict__ t1, const float* __restrict__ score,
                              const float* __restrict__ eemb, float* __restrict__ anchors2) {
  __shared__ int   e1s[16];
  __shared__ int   e2s[256];
  __shared__ float sc[256];
  __shared__ float red2[2];
  __shared__ float wp[8 * 128];
  int b = blockIdx.x, tid = threadIdx.x, lane = tid & 31, wid = tid >> 5;

  // ---- iter 0: N = 16 ----
  if (tid < 16) {
    e1s[tid] = ents1w[b * 16 + tid];
    int rel = adj_relation[(size_t)entity_idx[b] * 16 + tid];
    sc[tid] = score[(size_t)b * 32 + rel];
  }
  __syncthreads();
  if (tid == 0) {
    float m = -1e30f;
    for (int k = 0; k < 16; k++) m = fmaxf(m, sc[k]);
    float s = 0.f;
    for (int k = 0; k < 16; k++) { float e = __expf(sc[k] - m); sc[k] = e; s += e; }
    for (int k = 0; k < 16; k++) sc[k] /= s;
  }
  __syncthreads();
  if (tid < 128) {
    float acc = 0.f;
    for (int k = 0; k < 16; k++) acc += sc[k] * t1[((size_t)b * 16 + k) * 128 + tid];
    anchors2[(size_t)b * 128 + tid] = acc;          // s0
  }
  __syncthreads();

  // ---- iter 1: N = 256 ----
  {
    int k = tid >> 4, k2 = tid & 15;
    int e1 = e1s[k];
    e2s[tid] = adj_entity[(size_t)e1 * 16 + k2];
    int rel = adj_relation[(size_t)e1 * 16 + k2];
    sc[tid] = score[(size_t)(2048 + b) * 32 + rel];
  }
  __syncthreads();
  if (tid == 0) {
    float m = -1e30f;
    for (int n = 0; n < 256; n++) m = fmaxf(m, sc[n]);
    float s = 0.f;
    for (int n = 0; n < 256; n++) { float e = __expf(sc[n] - m); sc[n] = e; s += e; }
    red2[0] = s;
  }
  __syncthreads();
  float inv_s = 1.f / red2[0];
  float ax = 0.f, ay = 0.f, az = 0.f, aw = 0.f;
  for (int r = 0; r < 32; r++) {
    int n = wid * 32 + r;
    const float* row = eemb + (size_t)e2s[n] * 128;
    float4 q = *(const float4*)(row + lane * 4);
    float ss = q.x * q.x + q.y * q.y + q.z * q.z + q.w * q.w;
    ss = wave_sum32(ss);
    float wf = sc[n] * inv_s * renorm_fac(ss);
    ax += q.x * wf; ay += q.y * wf; az += q.z * wf; aw += q.w * wf;
  }
  wp[wid * 128 + lane * 4 + 0] = ax;
  wp[wid * 128 + lane * 4 + 1] = ay;
  wp[wid * 128 + lane * 4 + 2] = az;
  wp[wid * 128 + lane * 4 + 3] = aw;
  __syncthreads();
  if (tid < 128) {
    float s = 0.f;
    for (int w8 = 0; w8 < 8; w8++) s += wp[w8 * 128 + tid];
    anchors2[(size_t)(2048 + b) * 128 + tid] = s;   // s1w
  }
}

// ---------- P = anchor_exp + V, Q = anchor_exp * V ----------
__global__ void k_pq(const float* __restrict__ anchors, const float* __restrict__ V,
                     float* __restrict__ P, float* __restrict__ Q) {
  int idx = blockIdx.x * 256 + threadIdx.x;        // 2*2048*256 total, exact grid
  int i = idx >> 19;
  int rem = idx & 524287;
  int b = rem >> 8, j = rem & 255;
  float an = anchors[(size_t)(i * 2048 + b) * 128 + (j & 127)];
  float v = V[idx];
  P[idx] = an + v;
  Q[idx] = an * v;
}

extern "C" void kernel_launch(void* const* d_in, const int* in_sizes, int n_in,
                              void* d_out, int out_size, void* d_ws, size_t ws_size,
                              hipStream_t stream) {
  (void)in_sizes; (void)n_in; (void)out_size; (void)ws_size;
  const int*   entity_idx    = (const int*)d_in[0];
  const int*   adj_entity    = (const int*)d_in[1];
  const int*   adj_relation  = (const int*)d_in[2];
  const float* entity_embs   = (const float*)d_in[3];
  const float* relation_embs = (const float*)d_in[4];
  const float* A1   = (const float*)d_in[5];
  const float* A2   = (const float*)d_in[6];
  const float* A3   = (const float*)d_in[7];
  const float* Wx_w = (const float*)d_in[8];
  const float* Wx_b = (const float*)d_in[9];
  const float* W1_w = (const float*)d_in[10];
  const float* W1_b = (const float*)d_in[11];
  const float* W2_w = (const float*)d_in[12];
  const float* W2_b = (const float*)d_in[13];
  float* out = (float*)d_out;

  char* ws = (char*)d_ws;
  size_t off = 0;
  auto alloc = [&](size_t bytes) -> char* {
    char* p = ws + off;
    off += (bytes + 255) & ~(size_t)255;
    return p;
  };
  float*     t1       = (float*)alloc((size_t)2048 * 16 * 128 * 4);  // renormed hop-1 rows
  float*     anchors  = (float*)alloc((size_t)4096 * 128 * 4);       // [h ; s1sum]
  float*     anchors2 = (float*)alloc((size_t)4096 * 128 * 4);       // [s0 ; s1w]
  float*     ap       = (float*)alloc((size_t)4096 * 128 * 4);       // A1_left @ anchor
  float*     rproj    = (float*)alloc((size_t)32 * 128 * 4);
  float*     score    = (float*)alloc((size_t)131072 * 4);           // [iter][b][rel]
  int*       ents1    = (int*)alloc((size_t)2048 * 16 * 4);
  float*     V        = (float*)alloc((size_t)2 * 2048 * 256 * 4);
  float*     P        = (float*)alloc((size_t)2 * 2048 * 256 * 4);
  float*     Q        = (float*)alloc((size_t)2 * 2048 * 256 * 4);
  _Float16*  A1h      = (_Float16*)alloc((size_t)128 * 256 * 2);
  _Float16*  A2h      = (_Float16*)alloc((size_t)128 * 128 * 2);
  _Float16*  Wxh      = (_Float16*)alloc((size_t)2 * 256 * 128 * 2);
  _Float16*  W1h      = (_Float16*)alloc((size_t)256 * 256 * 2);
  _Float16*  W2h      = (_Float16*)alloc((size_t)256 * 256 * 2);

  // f32 -> f16 weight conversion
  k_cvt<<<(32768 + 255) / 256, 256, 0, stream>>>(A1, A1h, 32768);
  k_cvt<<<(16384 + 255) / 256, 256, 0, stream>>>(A2, A2h, 16384);
  k_cvt<<<(65536 + 255) / 256, 256, 0, stream>>>(Wx_w, Wxh, 65536);
  k_cvt<<<(65536 + 255) / 256, 256, 0, stream>>>(W1_w, W1h, 65536);
  k_cvt<<<(65536 + 255) / 256, 256, 0, stream>>>(W2_w, W2h, 65536);

  // relation projections (32 rows, exact fp32)
  k_rproj<<<32, 128, 0, stream>>>(relation_embs, A1, rproj);

  // gather + renorm (h, t1, s1sum) and h -> output cols 512..639
  k0_gather<<<2048, 128, 0, stream>>>(entity_idx, adj_entity, entity_embs, anchors, t1, ents1, out);

  // ap = anchors @ A1_left^T   (M=4096, N=128, K=128; A1 row stride 256)
  gemm16<<<dim3(256, 2), 128, 0, stream>>>(anchors, 128, A1h, 256, nullptr, ap, 128, 0, 128, 0, 0);

  // score table: 131072 rows -> 8192 tiles
  k_score<<<8192, 128, 0, stream>>>(ap, rproj, A2h, A3, score);

  // per-b softmax + weighted sums -> anchors2
  k4_softmax_ws<<<2048, 256, 0, stream>>>(entity_idx, adj_entity, adj_relation, ents1, t1, score,
                                          entity_embs, anchors2);

  // V_i = leaky(S_i @ Wxcat_i^T + Wx_b[i])  (M=2048, N=256, K=128)
  for (int i = 0; i < 2; i++) {
    gemm16<<<dim3(128, 4), 128, 0, stream>>>(anchors2 + (size_t)i * 2048 * 128, 128,
                                             Wxh + (size_t)i * 256 * 128, 128, Wx_b + i * 256,
                                             V + (size_t)i * 524288, 256, 0, 128, 1, 0);
  }

  // P/Q elementwise
  k_pq<<<4096, 256, 0, stream>>>(anchors, V, P, Q);

  // e_i = leaky(P_i @ W1^T + b1) + leaky(Q_i @ W2^T + b2) -> out cols (i==1 ? 0 : 256)
  for (int i = 0; i < 2; i++) {
    int col0 = (i == 0) ? 256 : 0;
    gemm16<<<dim3(128, 4), 128, 0, stream>>>(P + (size_t)i * 524288, 256, W1h, 256, W1_b, out, 640,
                                             col0, 256, 1, 0);
    gemm16<<<dim3(128, 4), 128, 0, stream>>>(Q + (size_t)i * 524288, 256, W2h, 256, W2_b, out, 640,
                                             col0, 256, 1, 1);
  }
}
